// Attention_6339371729466
// MI455X (gfx1250) — matrix operand order repaired and checked by reading, not compiled, not run on hardware
//
#include <hip/hip_runtime.h>
#include <hip/hip_bf16.h>
#include <stdint.h>

// ---------------------------------------------------------------------------
// Problem constants (match the reference)
// ---------------------------------------------------------------------------
#define DIMS      4096
#define NHEADS    32
#define NKVHEADS  8
#define HEADDIM   128
#define NREP      4
#define BATCH     4
#define SEQ       1024
#define TOKENS    (BATCH * SEQ)   // 4096

typedef __bf16 bf16_t;
typedef bf16_t v16bf __attribute__((ext_vector_type(16)));
typedef float  v8f   __attribute__((ext_vector_type(8)));

union Frag16 {
    v16bf          v;
    unsigned int   u[8];
    unsigned short s[16];
};

__device__ __forceinline__ unsigned short f2bf(float f) {
    unsigned int u = __float_as_uint(f);
    u = u + 0x7FFFu + ((u >> 16) & 1u);        // round-to-nearest-even
    return (unsigned short)(u >> 16);
}

__device__ __forceinline__ v8f v8f_zero() {
    v8f z;
#pragma unroll
    for (int i = 0; i < 8; ++i) z[i] = 0.0f;
    return z;
}

// A-fragment K permutation (16-bit, 16x32): dword pair p -> K start
__device__ __forceinline__ int a_kperm(int p, int half) {
    return 2 * p + 8 * half + ((p >= 4) ? 8 : 0);
}

// CDNA5 async global->LDS copy (16B per lane), tracked by ASYNCcnt.
// lds = byte offset of destination in LDS (low 32 bits of generic pointer).
__device__ __forceinline__ void async_b128(const void* g, unsigned lds) {
    asm volatile("global_load_async_to_lds_b128 %0, %1, off"
                 :: "v"(lds), "v"(g)
                 : "memory");
}
__device__ __forceinline__ void wait_async0() {
    asm volatile("s_wait_asynccnt 0x0" ::: "memory");
}
__device__ __forceinline__ void wait_async6() {
    asm volatile("s_wait_asynccnt 0x6" ::: "memory");
}

// ---------------------------------------------------------------------------
// f32 -> bf16 conversion (grid-stride)
// ---------------------------------------------------------------------------
__global__ __launch_bounds__(256) void f32_to_bf16_kernel(
    const float* __restrict__ in, unsigned short* __restrict__ out, long long n) {
    long long i = (long long)blockIdx.x * blockDim.x + threadIdx.x;
    long long stride = (long long)gridDim.x * blockDim.x;
    for (; i < n; i += stride) out[i] = f2bf(in[i]);
}

// ---------------------------------------------------------------------------
// RoPE: read f32 head-proj, apply rotation from freqs, write bf16
// layout: [TOKENS][nheads][HEADDIM]
// ---------------------------------------------------------------------------
__global__ __launch_bounds__(256) void rope_to_bf16_kernel(
    const float* __restrict__ h, const float* __restrict__ freqs,
    unsigned short* __restrict__ out, int nheads) {
    long long idx = (long long)blockIdx.x * blockDim.x + threadIdx.x;
    long long total = (long long)TOKENS * nheads * (HEADDIM / 2);
    if (idx >= total) return;
    int i        = (int)(idx % (HEADDIM / 2));
    long long th = idx / (HEADDIM / 2);
    int head     = (int)(th % nheads);
    long long t  = th / nheads;
    int pos      = (int)(t % SEQ);
    float f = freqs[pos * (HEADDIM / 2) + i];
    float c = __cosf(f), s = __sinf(f);
    size_t base = ((size_t)t * nheads + head) * HEADDIM + 2 * i;
    float a = h[base], b = h[base + 1];
    out[base]     = f2bf(a * c - b * s);
    out[base + 1] = f2bf(a * s + b * c);
}

// ---------------------------------------------------------------------------
// NT GEMM: C[M,N] (f32) = A[M,K] (bf16, row-major) x B[N,K]^T (bf16, row-major)
// 256 threads = 8 waves; block tile 256x128; wave tile 64x64 (4x4 WMMA tiles).
// Double-buffered LDS fed by GLOBAL_LOAD_ASYNC_TO_LDS_B128 (ASYNCcnt pipeline):
// every thread issues exactly 6 async b128 copies per K-slab (uniform per wave).
// ---------------------------------------------------------------------------
#define GBM 256
#define GBN 128
#define GBK 32
#define GLD (GBK + 8)   // ushort stride = 40 (80B, keeps 16B alignment, kills conflicts)

__global__ __launch_bounds__(256) void gemm_nt_bf16_wmma(
    const unsigned short* __restrict__ A,
    const unsigned short* __restrict__ B,
    float* __restrict__ C,
    int M, int N, int K) {
    __shared__ __align__(16) unsigned short As[2][GBM * GLD];
    __shared__ __align__(16) unsigned short Bs[2][GBN * GLD];

    const int tid  = threadIdx.x;
    const int lane = tid & 31;
    const int wid  = tid >> 5;
    const int wrow = wid >> 1;        // 0..3 -> 64 rows each
    const int wcol = wid & 1;         // 0..1 -> 64 cols each
    const int half = lane >> 4;
    const int l16  = lane & 15;

    const int m0 = blockIdx.y * GBM;
    const int n0 = blockIdx.x * GBN;

    // staging assignment: A row = tid (4 x 16B); B half-row (2 x 16B) -> 6/thread
    const int arow = tid;              // 0..255
    const int brow = tid >> 1;         // 0..127
    const int bcol = (tid & 1) << 4;   // 0 or 16 ushorts

    v8f acc[4][4];
#pragma unroll
    for (int mi = 0; mi < 4; ++mi)
#pragma unroll
        for (int nj = 0; nj < 4; ++nj) acc[mi][nj] = v8f_zero();

    const int nk = K / GBK;

    // prologue: issue K-slab 0 into buffer 0
    {
        const unsigned short* ga = A + (size_t)(m0 + arow) * K;
        unsigned la = (unsigned)(size_t)(&As[0][arow * GLD]);
        async_b128(ga,      la);
        async_b128(ga + 8,  la + 16);
        async_b128(ga + 16, la + 32);
        async_b128(ga + 24, la + 48);
        const unsigned short* gb = B + (size_t)(n0 + brow) * K + bcol;
        unsigned lb = (unsigned)(size_t)(&Bs[0][brow * GLD + bcol]);
        async_b128(gb,     lb);
        async_b128(gb + 8, lb + 16);
    }

    for (int kt = 0; kt < nk; ++kt) {
        const int p = kt & 1;
        if (kt + 1 < nk) {
            const int q = p ^ 1;
            const unsigned short* ga = A + (size_t)(m0 + arow) * K + (kt + 1) * GBK;
            unsigned la = (unsigned)(size_t)(&As[q][arow * GLD]);
            async_b128(ga,      la);
            async_b128(ga + 8,  la + 16);
            async_b128(ga + 16, la + 32);
            async_b128(ga + 24, la + 48);
            const unsigned short* gb = B + (size_t)(n0 + brow) * K + (kt + 1) * GBK + bcol;
            unsigned lb = (unsigned)(size_t)(&Bs[q][brow * GLD + bcol]);
            async_b128(gb,     lb);
            async_b128(gb + 8, lb + 16);
            wait_async6();   // <=6 outstanding: slab kt fully landed
        } else {
            wait_async0();
        }
        __syncthreads();     // all waves' copies for slab kt visible

        const unsigned short* asp = As[p];
        const unsigned short* bsp = Bs[p];
        Frag16 af[4], bfr[4];
#pragma unroll
        for (int mi = 0; mi < 4; ++mi) {
            int row = wrow * 64 + mi * 16 + l16;
#pragma unroll
            for (int p8 = 0; p8 < 8; ++p8)
                af[mi].u[p8] = *(const unsigned int*)(&asp[row * GLD + a_kperm(p8, half)]);
        }
#pragma unroll
        for (int nj = 0; nj < 4; ++nj) {
            int col = wcol * 64 + nj * 16 + l16;
#pragma unroll
            for (int p8 = 0; p8 < 8; ++p8)   // B layout: K = a_kperm order (matches A)
                bfr[nj].u[p8] = *(const unsigned int*)(&bsp[col * GLD + a_kperm(p8, half)]);
        }
#pragma unroll
        for (int mi = 0; mi < 4; ++mi)
#pragma unroll
            for (int nj = 0; nj < 4; ++nj)
                acc[mi][nj] = __builtin_amdgcn_wmma_f32_16x16x32_bf16(
                    false, af[mi].v, false, bfr[nj].v, (short)0, acc[mi][nj], false, false);

        __syncthreads();     // readers done before buffer p is overwritten (slab kt+2)
    }

#pragma unroll
    for (int mi = 0; mi < 4; ++mi) {
        int rowbase = m0 + wrow * 64 + mi * 16 + 8 * half;
#pragma unroll
        for (int nj = 0; nj < 4; ++nj) {
            int col = n0 + wcol * 64 + nj * 16 + l16;
#pragma unroll
            for (int r = 0; r < 8; ++r)
                C[(size_t)(rowbase + r) * N + col] = acc[mi][nj][r];
        }
    }
}

// ---------------------------------------------------------------------------
// Flash attention (causal, GQA 4:1), one wave per (b, qhead, 32-row q tile)
// Q,K,V,Ctx bf16; online softmax; all matmuls via v_wmma_f32_16x16x32_bf16;
// K/V chunks staged into LDS with async-to-LDS copies.
// ---------------------------------------------------------------------------
#define KVLD (HEADDIM + 8)   // 136 ushorts per key row (272B, 16B aligned)

__global__ __launch_bounds__(32) void flash_attn_wmma(
    const unsigned short* __restrict__ Q,    // [TOKENS][NHEADS][128]
    const unsigned short* __restrict__ Kc,   // [TOKENS][NKVHEADS][128]
    const unsigned short* __restrict__ Vc,   // [TOKENS][NKVHEADS][128]
    unsigned short* __restrict__ Ctx) {      // [TOKENS][NHEADS][128]
    __shared__ __align__(16) unsigned short Ks[32 * KVLD];
    __shared__ __align__(16) unsigned short Vs[32 * KVLD];
    __shared__ __align__(16) unsigned short Ps[32 * 40];

    const int lane = threadIdx.x;
    const int half = lane >> 4;
    const int l16  = lane & 15;

    const int qt  = blockIdx.x;       // 32-row q tile index
    const int qh  = blockIdx.y;       // query head
    const int b   = blockIdx.z;
    const int kvh = qh >> 2;          // NREP = 4

    // Q fragments (A layout), kept live across the key loop
    Frag16 aq[2][4];
#pragma unroll
    for (int mi = 0; mi < 2; ++mi) {
        const size_t t = (size_t)b * SEQ + qt * 32 + mi * 16 + l16;
        const unsigned short* qp = Q + (t * NHEADS + qh) * HEADDIM;
#pragma unroll
        for (int kk = 0; kk < 4; ++kk)
#pragma unroll
            for (int p = 0; p < 8; ++p)
                aq[mi][kk].u[p] = *(const unsigned int*)(qp + kk * 32 + a_kperm(p, half));
    }

    v8f o[2][8];
#pragma unroll
    for (int mi = 0; mi < 2; ++mi)
#pragma unroll
        for (int nd = 0; nd < 8; ++nd) o[mi][nd] = v8f_zero();

    float mrow[2][8], lrow[2][8];
#pragma unroll
    for (int mi = 0; mi < 2; ++mi)
#pragma unroll
        for (int r = 0; r < 8; ++r) { mrow[mi][r] = -1e30f; lrow[mi][r] = 0.0f; }

    const float SCALE = 0.08838834764831845f;  // 1/sqrt(128)

    for (int sc = 0; sc <= qt; ++sc) {
        // stage 32 keys x 128 of K and V into LDS via async-to-LDS b128 copies
        {
            const size_t tk = (size_t)b * SEQ + sc * 32 + lane;
            const unsigned short* gk = Kc + (tk * NKVHEADS + kvh) * HEADDIM;
            const unsigned short* gv = Vc + (tk * NKVHEADS + kvh) * HEADDIM;
            unsigned lk = (unsigned)(size_t)(&Ks[lane * KVLD]);
            unsigned lv = (unsigned)(size_t)(&Vs[lane * KVLD]);
#pragma unroll
            for (int i = 0; i < 16; ++i) {
                async_b128(gk + i * 8, lk + i * 16);
                async_b128(gv + i * 8, lv + i * 16);
            }
            wait_async0();
        }
        __syncthreads();

        // S = Q K^T  (32x32, contraction over HEADDIM in 4 steps of 32)
        v8f s[2][2];
#pragma unroll
        for (int mi = 0; mi < 2; ++mi)
#pragma unroll
            for (int nj = 0; nj < 2; ++nj) s[mi][nj] = v8f_zero();
#pragma unroll
        for (int kk = 0; kk < 4; ++kk) {
            Frag16 bk[2];
#pragma unroll
            for (int nj = 0; nj < 2; ++nj)
#pragma unroll
                for (int p = 0; p < 8; ++p)
                    bk[nj].u[p] = *(const unsigned int*)(
                        &Ks[(nj * 16 + l16) * KVLD + kk * 32 + a_kperm(p, half)]);
#pragma unroll
            for (int mi = 0; mi < 2; ++mi)
#pragma unroll
                for (int nj = 0; nj < 2; ++nj)
                    s[mi][nj] = __builtin_amdgcn_wmma_f32_16x16x32_bf16(
                        false, aq[mi][kk].v, false, bk[nj].v, (short)0, s[mi][nj], false, false);
        }

        // scale + causal mask + online softmax (row reductions over 16 lanes)
        const bool diag = (sc == qt);
#pragma unroll
        for (int mi = 0; mi < 2; ++mi) {
#pragma unroll
            for (int r = 0; r < 8; ++r) {
                int qrow = qt * 32 + mi * 16 + 8 * half + r;
                float s0 = s[mi][0][r] * SCALE;
                float s1 = s[mi][1][r] * SCALE;
                if (diag) {
                    if (sc * 32 + l16 > qrow)      s0 = -1e30f;
                    if (sc * 32 + 16 + l16 > qrow) s1 = -1e30f;
                }
                float cmax = fmaxf(s0, s1);
#pragma unroll
                for (int off = 1; off < 16; off <<= 1)
                    cmax = fmaxf(cmax, __shfl_xor(cmax, off, 32));
                float mnew  = fmaxf(mrow[mi][r], cmax);
                float alpha = __expf(mrow[mi][r] - mnew);
                float p0 = __expf(s0 - mnew);
                float p1 = __expf(s1 - mnew);
                float rs = p0 + p1;
#pragma unroll
                for (int off = 1; off < 16; off <<= 1)
                    rs += __shfl_xor(rs, off, 32);
                lrow[mi][r] = lrow[mi][r] * alpha + rs;
                mrow[mi][r] = mnew;
#pragma unroll
                for (int nd = 0; nd < 8; ++nd) o[mi][nd][r] *= alpha;
                int prow = mi * 16 + 8 * half + r;
                Ps[prow * 40 + l16]      = f2bf(p0);
                Ps[prow * 40 + 16 + l16] = f2bf(p1);
            }
        }

        // P as A-fragments (through LDS: C layout -> A layout)
        Frag16 ap[2];
#pragma unroll
        for (int mi = 0; mi < 2; ++mi)
#pragma unroll
            for (int p = 0; p < 8; ++p)
                ap[mi].u[p] = *(const unsigned int*)(
                    &Ps[(mi * 16 + l16) * 40 + a_kperm(p, half)]);

        // O += P V   (V as B operand: lane = dcol, element e -> key a_kperm(e/2, half) + e%2)
#pragma unroll
        for (int nd = 0; nd < 8; ++nd) {
            Frag16 bv;
#pragma unroll
            for (int e = 0; e < 16; ++e)
                bv.s[e] = Vs[(a_kperm(e >> 1, half) + (e & 1)) * KVLD + nd * 16 + l16];
#pragma unroll
            for (int mi = 0; mi < 2; ++mi)
                o[mi][nd] = __builtin_amdgcn_wmma_f32_16x16x32_bf16(
                    false, ap[mi].v, false, bv.v, (short)0, o[mi][nd], false, false);
        }
        __syncthreads();
    }

    // normalize and write context (bf16) for the final projection
#pragma unroll
    for (int mi = 0; mi < 2; ++mi) {
#pragma unroll
        for (int r = 0; r < 8; ++r) {
            float inv = 1.0f / lrow[mi][r];
            size_t t = (size_t)b * SEQ + qt * 32 + mi * 16 + 8 * half + r;
            unsigned short* cp = Ctx + (t * NHEADS + qh) * HEADDIM;
#pragma unroll
            for (int nd = 0; nd < 8; ++nd)
                cp[nd * 16 + l16] = f2bf(o[mi][nd][r] * inv);
        }
    }
}

// ---------------------------------------------------------------------------
// Host-side orchestration
// ---------------------------------------------------------------------------
extern "C" void kernel_launch(void* const* d_in, const int* in_sizes, int n_in,
                              void* d_out, int out_size, void* d_ws, size_t ws_size,
                              hipStream_t stream) {
    (void)in_sizes; (void)n_in; (void)out_size; (void)ws_size;

    const float* x     = (const float*)d_in[0];
    const float* wq    = (const float*)d_in[1];
    const float* wk    = (const float*)d_in[2];
    const float* wv    = (const float*)d_in[3];
    const float* wo    = (const float*)d_in[4];
    const float* freqs = (const float*)d_in[5];
    // d_in[6] = additive causal mask (reimplemented in-kernel), d_in[7] = starting_pos (0)

    char* ws = (char*)d_ws;
    size_t off = 0;
    auto carve = [&](size_t bytes) {
        char* p = ws + off;
        off += (bytes + 255) & ~(size_t)255;
        return p;
    };

    const size_t T   = TOKENS;
    const size_t KVD = (size_t)NKVHEADS * HEADDIM;   // 1024

    unsigned short* xb  = (unsigned short*)carve(T * DIMS * 2);
    unsigned short* wqb = (unsigned short*)carve((size_t)DIMS * DIMS * 2);
    unsigned short* wkb = (unsigned short*)carve(KVD * DIMS * 2);
    unsigned short* wvb = (unsigned short*)carve(KVD * DIMS * 2);
    unsigned short* wob = (unsigned short*)carve((size_t)DIMS * DIMS * 2);
    float* hq = (float*)carve(T * DIMS * 4);
    float* hk = (float*)carve(T * KVD * 4);
    float* hv = (float*)carve(T * KVD * 4);
    unsigned short* qb = (unsigned short*)carve(T * DIMS * 2);
    unsigned short* kb = (unsigned short*)carve(T * KVD * 2);
    unsigned short* vb = (unsigned short*)carve(T * KVD * 2);
    unsigned short* ctxb = xb;   // reuse: x_bf16 is dead after the V projection

    dim3 cb(256);

    // f32 -> bf16
    f32_to_bf16_kernel<<<4096, cb, 0, stream>>>(x,  xb,  (long long)(T * DIMS));
    f32_to_bf16_kernel<<<4096, cb, 0, stream>>>(wq, wqb, (long long)DIMS * DIMS);
    f32_to_bf16_kernel<<<1024, cb, 0, stream>>>(wk, wkb, (long long)(KVD * DIMS));
    f32_to_bf16_kernel<<<1024, cb, 0, stream>>>(wv, wvb, (long long)(KVD * DIMS));
    f32_to_bf16_kernel<<<4096, cb, 0, stream>>>(wo, wob, (long long)DIMS * DIMS);

    // projections (C = A * W^T), f32 accumulate
    gemm_nt_bf16_wmma<<<dim3(DIMS / GBN, (int)T / GBM), 256, 0, stream>>>(
        xb, wqb, hq, (int)T, DIMS, DIMS);
    gemm_nt_bf16_wmma<<<dim3((int)KVD / GBN, (int)T / GBM), 256, 0, stream>>>(
        xb, wkb, hk, (int)T, (int)KVD, DIMS);
    gemm_nt_bf16_wmma<<<dim3((int)KVD / GBN, (int)T / GBM), 256, 0, stream>>>(
        xb, wvb, hv, (int)T, (int)KVD, DIMS);

    // RoPE (f32 in, bf16 out) + V downconvert
    {
        long long npq = (long long)T * NHEADS * (HEADDIM / 2);
        rope_to_bf16_kernel<<<(unsigned)((npq + 255) / 256), cb, 0, stream>>>(hq, freqs, qb, NHEADS);
        long long npk = (long long)T * NKVHEADS * (HEADDIM / 2);
        rope_to_bf16_kernel<<<(unsigned)((npk + 255) / 256), cb, 0, stream>>>(hk, freqs, kb, NKVHEADS);
    }
    f32_to_bf16_kernel<<<1024, cb, 0, stream>>>(hv, vb, (long long)(T * KVD));

    // causal flash attention, writes bf16 context
    flash_attn_wmma<<<dim3(SEQ / 32, NHEADS, BATCH), 32, 0, stream>>>(qb, kb, vb, ctxb);

    // output projection -> f32 d_out
    gemm_nt_bf16_wmma<<<dim3(DIMS / GBN, (int)T / GBM), 256, 0, stream>>>(
        ctxb, wob, (float*)d_out, (int)T, DIMS, DIMS);
}
